// DistributedMixtureOfExperts_82729660055635
// MI455X (gfx1250) — matrix operand order repaired and checked by reading, not compiled, not run on hardware
//
#include <hip/hip_runtime.h>

// ---------------------------------------------------------------------------
// MoE forward for MI455X (gfx1250, wave32, WMMA).
// Problem: B=8 T=2048 C=1024 H=4096 E=8 K(topk)=2 capacity=320.
// Grouped GEMMs run on v_wmma_f32_16x16x32_bf16 (bf16 inputs, f32 accum).
// A-tile staging uses GLOBAL_LOAD_ASYNC_TO_LDS_B128 when available.
// ---------------------------------------------------------------------------

typedef __bf16 bf16x16 __attribute__((ext_vector_type(16)));
typedef float  f32x8   __attribute__((ext_vector_type(8)));

union Frag { bf16x16 v; unsigned u[8]; };

// Async global->LDS copy path (ASYNCcnt) if the toolchain exposes it.
#define USE_ASYNC_LDS 0
#if defined(__has_builtin)
#if __has_builtin(__builtin_amdgcn_global_load_async_to_lds_b128) && \
    __has_builtin(__builtin_amdgcn_s_wait_asynccnt)
#undef USE_ASYNC_LDS
#define USE_ASYNC_LDS 1
#endif
#endif

#if USE_ASYNC_LDS
// Builtin signature (from hipcc diagnostic): (v4i __global*, v4i __shared*, Ii, Ii)
typedef int v4i_ __attribute__((vector_size(16)));
typedef __attribute__((address_space(1))) v4i_ gv4i;
typedef __attribute__((address_space(3))) v4i_ sv4i;
// Generic LDS pointers carry the LDS offset in their low 32 bits (aperture
// mapping truncates addr[31:0]); generic global pointers equal AS1 numerically.
__device__ __forceinline__ sv4i* to_lds4(void* p) {
    return (sv4i*)(unsigned)(unsigned long long)p;
}
__device__ __forceinline__ gv4i* to_glob4(const void* p) {
    return (gv4i*)(unsigned long long)p;
}
#endif

__device__ __forceinline__ unsigned f_as_u(float f) { union { float f; unsigned u; } x; x.f = f; return x.u; }
__device__ __forceinline__ float    u_as_f(unsigned u) { union { float f; unsigned u; } x; x.u = u; return x.f; }
__device__ __forceinline__ unsigned short f2bf(float f) {
    unsigned u = f_as_u(f);
    return (unsigned short)((u + 0x7FFFu + ((u >> 16) & 1u)) >> 16);  // RNE
}
__device__ __forceinline__ float bf2f(unsigned short h) { return u_as_f(((unsigned)h) << 16); }

// ------------------------- 1. Router: softmax + top-2 ----------------------
__global__ __launch_bounds__(256)
void moe_router(const float* __restrict__ x, const float* __restrict__ Wr,
                int* __restrict__ eflat, float* __restrict__ pflat, int C)
{
    const int lane = threadIdx.x & 31;
    const int wave = threadIdx.x >> 5;
    const int token = blockIdx.x * 8 + wave;          // [0, B*T)
    const float* xp = x + (size_t)token * C;

    float a0=0,a1=0,a2=0,a3=0,a4=0,a5=0,a6=0,a7=0;
    for (int i = 0; i < C / 32; ++i) {
        int c = i * 32 + lane;
        float xv = xp[c];
        const float4* w = (const float4*)(Wr + (size_t)c * 8);
        float4 w0 = w[0], w1 = w[1];
        a0 += xv * w0.x; a1 += xv * w0.y; a2 += xv * w0.z; a3 += xv * w0.w;
        a4 += xv * w1.x; a5 += xv * w1.y; a6 += xv * w1.z; a7 += xv * w1.w;
    }
    for (int off = 16; off >= 1; off >>= 1) {
        a0 += __shfl_xor(a0, off, 32); a1 += __shfl_xor(a1, off, 32);
        a2 += __shfl_xor(a2, off, 32); a3 += __shfl_xor(a3, off, 32);
        a4 += __shfl_xor(a4, off, 32); a5 += __shfl_xor(a5, off, 32);
        a6 += __shfl_xor(a6, off, 32); a7 += __shfl_xor(a7, off, 32);
    }
    if (lane == 0) {
        float l[8] = {a0,a1,a2,a3,a4,a5,a6,a7};
        float mx = l[0];
        #pragma unroll
        for (int e = 1; e < 8; ++e) mx = l[e] > mx ? l[e] : mx;
        float ex[8], sum = 0.f;
        #pragma unroll
        for (int e = 0; e < 8; ++e) { ex[e] = __expf(l[e] - mx); sum += ex[e]; }
        int t1 = 0;
        #pragma unroll
        for (int e = 1; e < 8; ++e) if (l[e] > l[t1]) t1 = e;   // first-index ties
        int t2 = (t1 == 0) ? 1 : 0;
        #pragma unroll
        for (int e = 0; e < 8; ++e) if (e != t1 && l[e] > l[t2]) t2 = e;
        float inv = 1.f / sum;
        eflat[2 * token]     = t1;  pflat[2 * token]     = ex[t1] * inv;
        eflat[2 * token + 1] = t2;  pflat[2 * token + 1] = ex[t2] * inv;
    }
}

// ---------------- 2. Per-(batch,expert) exclusive positions ----------------
__global__ __launch_bounds__(32)
void moe_positions(const int* __restrict__ eflat, int* __restrict__ pos, int S)
{
    const int b = blockIdx.x;
    const int lane = threadIdx.x;
    const unsigned lt = (1u << lane) - 1u;
    int base[8] = {0,0,0,0,0,0,0,0};
    for (int chunk = 0; chunk < S; chunk += 32) {
        int e = eflat[b * S + chunk + lane];
        unsigned bal[8];
        #pragma unroll
        for (int ex = 0; ex < 8; ++ex) bal[ex] = (unsigned)__ballot(e == ex);
        int p = 0;
        #pragma unroll
        for (int ex = 0; ex < 8; ++ex) {
            int cand = base[ex] + __popc(bal[ex] & lt);
            if (e == ex) p = cand;
        }
        pos[b * S + chunk + lane] = p;
        #pragma unroll
        for (int ex = 0; ex < 8; ++ex) base[ex] += __popc(bal[ex]);
    }
}

// ----------------------------- 3a. Zero buf --------------------------------
__global__ __launch_bounds__(256)
void moe_zero(uint4* __restrict__ p)
{
    size_t i = (size_t)blockIdx.x * 256 + threadIdx.x;
    p[i] = make_uint4(0u, 0u, 0u, 0u);
}

// --------------------- 3b. Dispatch x -> bf16 buckets ----------------------
__global__ __launch_bounds__(256)
void moe_dispatch(const float* __restrict__ x, const int* __restrict__ eflat,
                  const int* __restrict__ pos, unsigned short* __restrict__ buf,
                  int S, int C, int cap, int Bsz)
{
    const int lane = threadIdx.x & 31;
    const int wave = threadIdx.x >> 5;
    const int g = blockIdx.x * 8 + wave;              // slot id in [0, B*S)
    const int b = g / S;
    const int s = g - b * S;
    const int t = s >> 1;
    int p = pos[g];
    if (p >= cap) return;                             // wave-uniform
    int e = eflat[g];
    size_t srow = ((size_t)b * (S >> 1) + t) * (size_t)C;
    size_t drow = (((size_t)e * Bsz + b) * cap + p) * (size_t)C;
    #pragma unroll
    for (int i = 0; i < 8; ++i) {
        int idx = i * 128 + lane * 4;
        float4 v = *(const float4*)(x + srow + idx);
        unsigned lo = (unsigned)f2bf(v.x) | ((unsigned)f2bf(v.y) << 16);
        unsigned hi = (unsigned)f2bf(v.z) | ((unsigned)f2bf(v.w) << 16);
        *(uint2*)(buf + drow + idx) = make_uint2(lo, hi);
    }
}

// --------------------- 4. Grouped GEMM on WMMA bf16 ------------------------
// Block: 128 threads (4 waves). Tile BM=256 x BN=64, K-step 32.
// Each wave owns 64 rows: 4 A-frags x 4 B-frags = 16 WMMA per K-step
// (16 ds_load_b128 per K-step -> 1.0 LDS loads per WMMA).
#define BM 256
#define BN 64
#define BK 32
#define LDA 40   // padded LDS row stride in shorts (80B, 16B aligned)
#define LDB 40

__global__ __launch_bounds__(128)
void moe_gemm(const unsigned short* __restrict__ A, const float* __restrict__ W,
              const float* __restrict__ bias, unsigned short* __restrict__ Out,
              int K, int N, int rowsPerExpert, int doRelu)
{
    __shared__ unsigned short As[BM * LDA];   // 20480 B
    __shared__ unsigned short Bs[BN * LDB];   //  5120 B

    const int e   = blockIdx.z;
    const int nt  = blockIdx.x * BN;
    const int rowbase = e * rowsPerExpert + blockIdx.y * BM;
    const int tid  = threadIdx.x;
    const int lane = tid & 31;
    const int wave = tid >> 5;
    const int half = lane >> 4;
    const int l16  = lane & 15;

    const float* We = W + (size_t)e * K * N;

    f32x8 acc[4][4];
    #pragma unroll
    for (int i = 0; i < 4; ++i)
        #pragma unroll
        for (int j = 0; j < 4; ++j)
            acc[i][j] = (f32x8){0.f,0.f,0.f,0.f,0.f,0.f,0.f,0.f};

    for (int kt = 0; kt < K; kt += BK) {
        // A tile: 256 rows x 32 bf16; thread stages rows 2*tid, 2*tid+1.
        {
            #pragma unroll
            for (int rr = 0; rr < 2; ++rr) {
                int row = tid * 2 + rr;
                const unsigned short* gp = A + (size_t)(rowbase + row) * K + kt;
                unsigned short* lp = As + row * LDA;
#if USE_ASYNC_LDS
                #pragma unroll
                for (int seg = 0; seg < 4; ++seg)
                    __builtin_amdgcn_global_load_async_to_lds_b128(
                        to_glob4(gp + seg * 8), to_lds4(lp + seg * 8), 0, 0);
#else
                const uint4* src = (const uint4*)gp;
                uint4* dst = (uint4*)lp;
                dst[0] = src[0]; dst[1] = src[1]; dst[2] = src[2]; dst[3] = src[3];
#endif
            }
        }
        // B tile: column n, 16 K-values each; f32 -> bf16, n-major in LDS.
        {
            int n  = tid & 63;
            int kg = tid >> 6;                         // 0..1
            const float* src = We + (size_t)(kt + kg * 16) * N + nt + n;
            unsigned pk[8];
            #pragma unroll
            for (int k2 = 0; k2 < 16; k2 += 2) {
                float v0 = src[(size_t)k2 * N];
                float v1 = src[(size_t)(k2 + 1) * N];
                pk[k2 >> 1] = (unsigned)f2bf(v0) | ((unsigned)f2bf(v1) << 16);
            }
            uint4* dst = (uint4*)(Bs + n * LDB + kg * 16);
            dst[0] = make_uint4(pk[0], pk[1], pk[2], pk[3]);
            dst[1] = make_uint4(pk[4], pk[5], pk[6], pk[7]);
        }
#if USE_ASYNC_LDS
        __builtin_amdgcn_s_wait_asynccnt(0);
#endif
        __syncthreads();

        // A fragments (16x32 layout: lanes 0-15 K={0..7,16..23}, 16-31 +8).
        Frag afr[4];
        #pragma unroll
        for (int i = 0; i < 4; ++i) {
            const unsigned short* ap = As + (wave * 64 + i * 16 + l16) * LDA;
            #pragma unroll
            for (int v = 0; v < 8; ++v) {
                int k0 = (v < 4) ? (2 * v + 8 * half) : (16 + 2 * (v - 4) + 8 * half);
                afr[i].u[v] = *(const unsigned*)(ap + k0);
            }
        }
        // B fragments + WMMA. B 32x16: lanes 0-15 K={0..7,16..23}, 16-31 +8.
        #pragma unroll
        for (int j = 0; j < 4; ++j) {
            Frag bfr;
            const unsigned short* bp = Bs + (j * 16 + l16) * LDB;
            #pragma unroll
            for (int v = 0; v < 8; ++v) bfr.u[v] = *(const unsigned*)(bp + ((v < 4) ? (2 * v + 8 * half) : (16 + 2 * (v - 4) + 8 * half)));
            #pragma unroll
            for (int i = 0; i < 4; ++i)
                acc[i][j] = __builtin_amdgcn_wmma_f32_16x16x32_bf16(
                    false, afr[i].v, false, bfr.v, (short)0, acc[i][j], false, false);
        }
        __syncthreads();
    }

    // Epilogue: C/D layout lanes 0-15: N=lane, M=r; lanes 16-31: M=r+8.
    #pragma unroll
    for (int i = 0; i < 4; ++i) {
        int mrow = rowbase + wave * 64 + i * 16 + half * 8;
        #pragma unroll
        for (int j = 0; j < 4; ++j) {
            int col = nt + j * 16 + l16;
            float bv = bias[(size_t)e * N + col];
            #pragma unroll
            for (int r = 0; r < 8; ++r) {
                float v = acc[i][j][r] + bv;
                if (doRelu) v = v > 0.f ? v : 0.f;
                Out[(size_t)(mrow + r) * N + col] = f2bf(v);
            }
        }
    }
}

// --------------------------- 5. Combine ------------------------------------
__global__ __launch_bounds__(256)
void moe_combine(const unsigned short* __restrict__ y, const int* __restrict__ eflat,
                 const float* __restrict__ pflat, const int* __restrict__ pos,
                 float* __restrict__ out, int C, int cap, int Bsz, int T)
{
    const int token = blockIdx.x;                     // b*T + t
    const int b = token / T;
    const int g0 = token * 2, g1 = g0 + 1;
    int e0 = eflat[g0], e1 = eflat[g1];
    int p0 = pos[g0],  p1 = pos[g1];
    float w0 = (p0 < cap) ? pflat[g0] : 0.f;
    float w1 = (p1 < cap) ? pflat[g1] : 0.f;
    int q0 = (p0 < cap) ? p0 : cap - 1;
    int q1 = (p1 < cap) ? p1 : cap - 1;
    size_t r0 = (((size_t)e0 * Bsz + b) * cap + q0) * (size_t)C;
    size_t r1 = (((size_t)e1 * Bsz + b) * cap + q1) * (size_t)C;
    int c = threadIdx.x * 4;
    uint2 ya = *(const uint2*)(y + r0 + c);
    uint2 yb = *(const uint2*)(y + r1 + c);
    float4 o;
    o.x = w0 * bf2f((unsigned short)(ya.x & 0xffffu)) + w1 * bf2f((unsigned short)(yb.x & 0xffffu));
    o.y = w0 * bf2f((unsigned short)(ya.x >> 16))     + w1 * bf2f((unsigned short)(yb.x >> 16));
    o.z = w0 * bf2f((unsigned short)(ya.y & 0xffffu)) + w1 * bf2f((unsigned short)(yb.y & 0xffffu));
    o.w = w0 * bf2f((unsigned short)(ya.y >> 16))     + w1 * bf2f((unsigned short)(yb.y >> 16));
    *(float4*)(out + (size_t)token * C + c) = o;
}

// ---------------------------------------------------------------------------
extern "C" void kernel_launch(void* const* d_in, const int* in_sizes, int n_in,
                              void* d_out, int out_size, void* d_ws, size_t ws_size,
                              hipStream_t stream)
{
    (void)in_sizes; (void)n_in; (void)out_size; (void)ws_size;
    const int B = 8, T = 2048, C = 1024, H = 4096, E = 8;
    const int S = 2 * T;                 // T * topk
    const int cap = 320;                 // int(T/E * 1.25)
    const int RPE = B * cap;             // 2560 rows per expert

    const float* x  = (const float*)d_in[0];
    const float* Wr = (const float*)d_in[1];
    const float* W1 = (const float*)d_in[2];
    const float* b1 = (const float*)d_in[3];
    const float* W2 = (const float*)d_in[4];
    const float* b2 = (const float*)d_in[5];
    float* out = (float*)d_out;

    // workspace layout (bytes)
    char* ws = (char*)d_ws;
    const size_t eflat_off = 0;                                  // B*S*4   = 131072
    const size_t pflat_off = 131072;                             // B*S*4
    const size_t pos_off   = 262144;                             // B*S*4
    const size_t buf_off   = 393216;                             // E*B*cap*C*2 = 41943040
    const size_t h_off     = buf_off + 41943040ull;              // E*B*cap*H*2 = 167772160
    const size_t y_off     = h_off + 167772160ull;               // E*B*cap*C*2 = 41943040

    int*   eflat = (int*)(ws + eflat_off);
    float* pflat = (float*)(ws + pflat_off);
    int*   pos   = (int*)(ws + pos_off);
    unsigned short* buf = (unsigned short*)(ws + buf_off);
    unsigned short* h   = (unsigned short*)(ws + h_off);
    unsigned short* y   = (unsigned short*)(ws + y_off);

    // 1. router: B*T tokens, 8 waves / block
    moe_router<<<(B * T) / 8, 256, 0, stream>>>(x, Wr, eflat, pflat, C);
    // 2. positions: one wave per batch row
    moe_positions<<<B, 32, 0, stream>>>(eflat, pos, S);
    // 3. zero dispatch buffer (41943040 B = 2621440 uint4 = 10240 * 256)
    moe_zero<<<10240, 256, 0, stream>>>((uint4*)buf);
    //    dispatch: one wave per slot
    moe_dispatch<<<(B * S) / 8, 256, 0, stream>>>(x, eflat, pos, buf, S, C, cap, B);
    // 4. grouped GEMMs (WMMA bf16): h = relu(buf@W1+b1); y = h@W2+b2
    moe_gemm<<<dim3(H / BN, RPE / BM, E), 128, 0, stream>>>(buf, W1, b1, h, C, H, RPE, 1);
    moe_gemm<<<dim3(C / BN, RPE / BM, E), 128, 0, stream>>>(h,  W2, b2, y, H, C, RPE, 0);
    // 5. combine
    moe_combine<<<B * T, 256, 0, stream>>>(y, eflat, pflat, pos, out, C, cap, B, T);
}
